// MaskedSelfAttention_13683765805854
// MI455X (gfx1250) — matrix operand order repaired and checked by reading, not compiled, not run on hardware
//
#include <hip/hip_runtime.h>

typedef __bf16 bf16_t;
typedef __bf16 v16bf __attribute__((ext_vector_type(16)));
typedef float  v8f   __attribute__((ext_vector_type(8)));
typedef int    v4i   __attribute__((vector_size(16)));

#define B_  8
#define S_  2048
#define E_  1024
#define H_  128

#if __has_builtin(__builtin_amdgcn_global_load_async_to_lds_b128)
#define HAVE_ASYNC_LDS 1
typedef __attribute__((address_space(1))) v4i* gv4i_ptr;
typedef __attribute__((address_space(3))) v4i* lv4i_ptr;
#else
#define HAVE_ASYNC_LDS 0
#endif

#if __has_builtin(__builtin_amdgcn_tensor_load_to_lds) && \
    __has_builtin(__builtin_amdgcn_s_wait_tensorcnt)
#define HAVE_TDM 1
typedef unsigned int v4u_e __attribute__((ext_vector_type(4)));
typedef int          v8i_e __attribute__((ext_vector_type(8)));
typedef int          v4i_e __attribute__((ext_vector_type(4)));
typedef __attribute__((address_space(3))) char* lchar_ptr;
#else
#define HAVE_TDM 0
#endif

// ---------------------------------------------------------------------------
// Kernel 0: W (E,H) fp32  ->  W^T (H,E) bf16.  Makes every WMMA B-fragment a
// contiguous 32-byte load (column of W == row of W^T).  3*256 KB, L2-resident.
// ---------------------------------------------------------------------------
__global__ __launch_bounds__(256) void w_transpose_kernel(
    const float* __restrict__ Wq, const float* __restrict__ Wk,
    const float* __restrict__ Wv,
    bf16_t* __restrict__ Wqt, bf16_t* __restrict__ Wkt, bf16_t* __restrict__ Wvt)
{
  const int idx = blockIdx.x * 256 + threadIdx.x;   // over E*H, reads coalesced
  if (idx >= E_ * H_) return;
  const int e = idx >> 7;           // idx / H
  const int n = idx & (H_ - 1);     // idx % H
  Wqt[(size_t)n * E_ + e] = (bf16_t)Wq[idx];
  Wkt[(size_t)n * E_ + e] = (bf16_t)Wk[idx];
  Wvt[(size_t)n * E_ + e] = (bf16_t)Wv[idx];
}

// ---------------------------------------------------------------------------
// Kernel 1: Q/K/V projection.  (B*S, E) @ (E, H) + bias -> bf16 workspace.
// Block = 256 threads = 8 waves; block owns a 16-row M-tile, wave w owns
// columns [16w, 16w+16).  E=1024 -> 32 WMMA K-steps of 32.
// ---------------------------------------------------------------------------
__global__ __launch_bounds__(256) void qkv_proj_kernel(
    const float* __restrict__ x,
    const bf16_t* __restrict__ Wqt, const float* __restrict__ bq,
    const bf16_t* __restrict__ Wkt, const float* __restrict__ bk,
    const bf16_t* __restrict__ Wvt, const float* __restrict__ bv,
    bf16_t* __restrict__ qo, bf16_t* __restrict__ ko, bf16_t* __restrict__ vo)
{
  __shared__ __align__(16) bf16_t xs[16][E_];   // 32 KB x-tile as bf16

  const int tid  = threadIdx.x;
  const int lane = tid & 31;
  const int wave = tid >> 5;
  const int row0 = blockIdx.x * 16;             // flat (b*S + s) row base

  // stage x tile 16 x 1024 (fp32 -> bf16), coalesced across threads
  for (int i = tid; i < 16 * E_; i += 256) {
    const int r = i >> 10, c = i & (E_ - 1);
    xs[r][c] = (bf16_t)x[(size_t)(row0 + r) * E_ + c];
  }
  __syncthreads();

  // A-frag (16x32 bf16, contiguous k order, same as B): lane l -> M = l&15; lanes>=16 hold K 16..31
  const int mrow = lane & 15;
  const int aoff = (lane >> 4) * 16;            // 0 or 16
  const int boff = (lane >> 4) * 16;            // B-frag K-row base: 0 or 16
  const int n    = wave * 16 + (lane & 15);     // output column

  // per-lane row of W^T = column n of W; fragment = 32 contiguous bytes
  const bf16_t* wqrow = Wqt + (size_t)n * E_ + boff;
  const bf16_t* wkrow = Wkt + (size_t)n * E_ + boff;
  const bf16_t* wvrow = Wvt + (size_t)n * E_ + boff;

  v8f accq = {}, acck = {}, accv = {};

  for (int kk = 0; kk < E_ / 32; ++kk) {
    const int kbase = kk * 32;
    v16bf a;
#pragma unroll
    for (int j = 0; j < 8; ++j) {
      a[j]     = xs[mrow][kbase + aoff + j];
      a[j + 8] = xs[mrow][kbase + aoff + 8 + j];
    }
    const v16bf fq = *(const v16bf*)(wqrow + kbase);
    const v16bf fk = *(const v16bf*)(wkrow + kbase);
    const v16bf fv = *(const v16bf*)(wvrow + kbase);
    accq = __builtin_amdgcn_wmma_f32_16x16x32_bf16(false, a, false, fq, (short)0, accq, false, false);
    acck = __builtin_amdgcn_wmma_f32_16x16x32_bf16(false, a, false, fk, (short)0, acck, false, false);
    accv = __builtin_amdgcn_wmma_f32_16x16x32_bf16(false, a, false, fv, (short)0, accv, false, false);
  }

  // bias + store (C layout: VGPR r, lane l -> M = r + 8*(l>>4), N = l&15)
  const float vbq = bq[n], vbk = bk[n], vbv = bv[n];
  const int mbase = (lane >> 4) * 8;
#pragma unroll
  for (int r = 0; r < 8; ++r) {
    const size_t grow = (size_t)(row0 + mbase + r) * H_ + n;
    qo[grow] = (bf16_t)(accq[r] + vbq);
    ko[grow] = (bf16_t)(acck[r] + vbk);
    vo[grow] = (bf16_t)(accv[r] + vbv);
  }
}

// ---------------------------------------------------------------------------
// Kernel 2: masked flash attention.  Block = 128 threads = 4 waves, each wave
// owns 16 query rows with the full 16x128 fp32 accumulator.  Keys processed
// in chunks of 32.  K chunk staged via Tensor Data Mover (1-D tile of 4096
// bf16, contiguous), V chunk via async-to-LDS; both shared by the 4 waves.
// ---------------------------------------------------------------------------
__global__ __launch_bounds__(128) void attn_kernel(
    const bf16_t* __restrict__ q,
    const bf16_t* __restrict__ k,
    const bf16_t* __restrict__ v,
    const int*    __restrict__ mask,
    float*        __restrict__ out)
{
  __shared__ __align__(16) bf16_t Ks[32 * H_];  // 8 KB
  __shared__ __align__(16) bf16_t Vs[32 * H_];  // 8 KB
  __shared__ __align__(16) bf16_t Ps[4][16][32];// per-wave P relayout, 4 KB

  const int tid  = threadIdx.x;
  const int lane = tid & 31;
  const int wave = tid >> 5;
  const int b    = blockIdx.x >> 5;             // 32 blocks per batch
  const int tile = blockIdx.x & 31;
  const int q0   = tile * 64 + wave * 16;       // query base in this batch

  const int mrow = lane & 15;
  const int half = lane >> 4;
  const int aoff = half * 16;

  // Q A-frags: 16 queries x 128 dims -> 4 frags of 16x32
  v16bf qa[4];
#pragma unroll
  for (int kk = 0; kk < 4; ++kk) {
    const bf16_t* qrow = q + (size_t)(b * S_ + q0 + mrow) * H_ + kk * 32;
#pragma unroll
    for (int j = 0; j < 8; ++j) {
      qa[kk][j]     = qrow[aoff + j];
      qa[kk][j + 8] = qrow[aoff + 8 + j];
    }
  }

  v8f o[8];
#pragma unroll
  for (int t = 0; t < 8; ++t) o[t] = (v8f){};
  float mrun[8], lrun[8];
#pragma unroll
  for (int r = 0; r < 8; ++r) { mrun[r] = -3.0e38f; lrun[r] = 0.0f; }

  const float scale = 0.08838834764831845f;     // 1/sqrt(128)

#if HAVE_TDM
  const unsigned int ksLds = (unsigned int)(size_t)(lchar_ptr)(void*)Ks;
#endif

  for (int c = 0; c < S_ / 32; ++c) {
    __syncthreads();                            // Ks/Vs safe to overwrite
    {
      const bf16_t* ksrc = k + (size_t)(b * S_ + c * 32) * H_;
      const bf16_t* vsrc = v + (size_t)(b * S_ + c * 32) * H_;

#if HAVE_TDM
      // ---- K chunk: one TDM descriptor, 1-D tile of 4096 bf16 (8 KB) ------
      if (wave == 0) {
        const unsigned long long ga = (unsigned long long)(size_t)ksrc;
        v4u_e g0;
        g0[0] = 1u;                                   // count=1, no gather
        g0[1] = ksLds;                                // lds_addr (bytes)
        g0[2] = (unsigned int)(ga & 0xFFFFFFFFu);     // global_addr[31:0]
        g0[3] = (unsigned int)((ga >> 32) & 0x01FFFFFFu) | (2u << 30); // type=2
        v8i_e g1;
        g1[0] = 0x00010000;        // workgroup_mask=0, data_size=1 (2 bytes)
        g1[1] = (int)(4096u << 16);// tensor_dim0[15:0]=4096 at bits 63:48
        g1[2] = (int)(1u << 16);   // tensor_dim0[31:16]=0, tensor_dim1=1
        g1[3] = (int)(4096u << 16);// tensor_dim1[31:16]=0, tile_dim0=4096
        g1[4] = 1;                 // tile_dim1=1, tile_dim2=0
        g1[5] = 4096;              // tensor_dim0_stride[31:0]
        g1[6] = 0;                 // stride0[47:32]=0, tensor_dim1_stride lo=0
        g1[7] = 0;
        const v4i_e gz4 = {0, 0, 0, 0};
        const v8i_e gz8 = {0, 0, 0, 0, 0, 0, 0, 0};
        __builtin_amdgcn_tensor_load_to_lds(g0, g1, gz4, gz4, gz8, 0);
        __builtin_amdgcn_s_wait_tensorcnt(0);
      }
#elif HAVE_ASYNC_LDS
      for (int i = tid; i < 512; i += 128) {
        __builtin_amdgcn_global_load_async_to_lds_b128(
            (gv4i_ptr)(ksrc + (size_t)i * 8),
            (lv4i_ptr)(Ks + (size_t)i * 8), 0, 0);
      }
#else
      {
        const uint4* k4 = (const uint4*)ksrc;
        uint4* kdst = (uint4*)Ks;
        for (int i = tid; i < 512; i += 128) kdst[i] = k4[i];
      }
#endif

#if HAVE_ASYNC_LDS
      // ---- V chunk: async copy memory -> LDS, ASYNCcnt-tracked ------------
      for (int i = tid; i < 512; i += 128) {    // 512 x b128
        __builtin_amdgcn_global_load_async_to_lds_b128(
            (gv4i_ptr)(vsrc + (size_t)i * 8),
            (lv4i_ptr)(Vs + (size_t)i * 8), 0, 0);
      }
#if __has_builtin(__builtin_amdgcn_s_wait_asynccnt)
      __builtin_amdgcn_s_wait_asynccnt(0);
#else
      asm volatile("s_wait_asynccnt 0x0" ::: "memory");
#endif
#else
      {
        const uint4* v4 = (const uint4*)vsrc;
        uint4* vdst = (uint4*)Vs;
        for (int i = tid; i < 512; i += 128) vdst[i] = v4[i];
      }
#endif
    }
    __syncthreads();

    // ---- S = Q @ K^T : two 16x16 tiles (keys 0-15, 16-31 of the chunk)
    v8f s0 = {}, s1 = {};
#pragma unroll
    for (int kk = 0; kk < 4; ++kk) {
      const bf16_t* k0 = Ks + (size_t)(lane & 15) * H_ + kk * 32 + half * 16;
      const v16bf fk0 = *(const v16bf*)k0;           // 32B aligned
      const v16bf fk1 = *(const v16bf*)(k0 + 16 * H_);
      s0 = __builtin_amdgcn_wmma_f32_16x16x32_bf16(false, qa[kk], false, fk0, (short)0, s0, false, false);
      s1 = __builtin_amdgcn_wmma_f32_16x16x32_bf16(false, qa[kk], false, fk1, (short)0, s1, false, false);
    }

    // ---- scale + key mask (N dim = key; lane holds one key per tile)
    const int key0 = c * 32 + (lane & 15);
    const int msk0 = mask[b * S_ + key0];
    const int msk1 = mask[b * S_ + key0 + 16];
#pragma unroll
    for (int r = 0; r < 8; ++r) {
      s0[r] = (msk0 == 0) ? -1.0e9f : s0[r] * scale;
      s1[r] = (msk1 == 0) ? -1.0e9f : s1[r] * scale;
    }

    // ---- online softmax (row r lives in lanes of the same 16-lane half)
    float alpha[8];
#pragma unroll
    for (int r = 0; r < 8; ++r) {
      float rm = fmaxf(s0[r], s1[r]);
      rm = fmaxf(rm, __shfl_xor(rm, 1, 32));
      rm = fmaxf(rm, __shfl_xor(rm, 2, 32));
      rm = fmaxf(rm, __shfl_xor(rm, 4, 32));
      rm = fmaxf(rm, __shfl_xor(rm, 8, 32));
      const float mnew = fmaxf(mrun[r], rm);
      alpha[r] = __expf(mrun[r] - mnew);
      const float p0 = __expf(s0[r] - mnew);
      const float p1 = __expf(s1[r] - mnew);
      s0[r] = p0; s1[r] = p1;
      float rs = p0 + p1;
      rs += __shfl_xor(rs, 1, 32);
      rs += __shfl_xor(rs, 2, 32);
      rs += __shfl_xor(rs, 4, 32);
      rs += __shfl_xor(rs, 8, 32);
      lrun[r] = lrun[r] * alpha[r] + rs;
      mrun[r] = mnew;
    }

    // ---- rescale accumulator
#pragma unroll
    for (int t = 0; t < 8; ++t)
#pragma unroll
      for (int r = 0; r < 8; ++r) o[t][r] *= alpha[r];

    // ---- C-layout P -> row-major LDS -> A-layout frag
#pragma unroll
    for (int r = 0; r < 8; ++r) {
      Ps[wave][r + half * 8][lane & 15]        = (bf16_t)s0[r];
      Ps[wave][r + half * 8][16 + (lane & 15)] = (bf16_t)s1[r];
    }
    __syncthreads();

    v16bf pa;
#pragma unroll
    for (int j = 0; j < 8; ++j) {
      pa[j]     = Ps[wave][mrow][aoff + j];
      pa[j + 8] = Ps[wave][mrow][aoff + 8 + j];
    }

    // ---- O += P @ V  (8 N-tiles covering H=128)
#pragma unroll
    for (int t = 0; t < 8; ++t) {
      v16bf fv;
      const int n = t * 16 + (lane & 15);
#pragma unroll
      for (int j = 0; j < 16; ++j) fv[j] = Vs[(size_t)(half * 16 + j) * H_ + n];
      o[t] = __builtin_amdgcn_wmma_f32_16x16x32_bf16(false, pa, false, fv, (short)0, o[t], false, false);
    }
  }

  // ---- normalize + store fp32
#pragma unroll
  for (int t = 0; t < 8; ++t) {
    const int n = t * 16 + (lane & 15);
#pragma unroll
    for (int r = 0; r < 8; ++r) {
      const int m = r + half * 8;
      out[(size_t)(b * S_ + q0 + m) * H_ + n] = o[t][r] / lrun[r];
    }
  }
}

// ---------------------------------------------------------------------------
extern "C" void kernel_launch(void* const* d_in, const int* in_sizes, int n_in,
                              void* d_out, int out_size, void* d_ws, size_t ws_size,
                              hipStream_t stream) {
  (void)in_sizes; (void)n_in; (void)out_size; (void)ws_size;
  const float* x    = (const float*)d_in[0];
  const int*   mask = (const int*)  d_in[1];
  const float* Wq   = (const float*)d_in[2];
  const float* bq   = (const float*)d_in[3];
  const float* Wk   = (const float*)d_in[4];
  const float* bk   = (const float*)d_in[5];
  const float* Wv   = (const float*)d_in[6];
  const float* bv   = (const float*)d_in[7];
  float* out = (float*)d_out;

  // workspace: bf16 Q,K,V (3 x 4 MB) + bf16 W^T x3 (3 x 256 KB)
  bf16_t* qws = (bf16_t*)d_ws;
  bf16_t* kws = qws + (size_t)B_ * S_ * H_;
  bf16_t* vws = kws + (size_t)B_ * S_ * H_;
  bf16_t* wqt = vws + (size_t)B_ * S_ * H_;
  bf16_t* wkt = wqt + (size_t)E_ * H_;
  bf16_t* wvt = wkt + (size_t)E_ * H_;

  w_transpose_kernel<<<(E_ * H_ + 255) / 256, 256, 0, stream>>>(
      Wq, Wk, Wv, wqt, wkt, wvt);
  qkv_proj_kernel<<<(B_ * S_) / 16, 256, 0, stream>>>(
      x, wqt, bq, wkt, bk, wvt, bv, qws, kws, vws);
  attn_kernel<<<(B_ * S_) / 64, 128, 0, stream>>>(
      qws, kws, vws, mask, out);
}
